// GATv2_69569880261281
// MI455X (gfx1250) — matrix operand order repaired and checked by reading, not compiled, not run on hardware
//
#include <hip/hip_runtime.h>
#include <hip/hip_bf16.h>
#include <math.h>

typedef __attribute__((ext_vector_type(16))) _Float16 v16h;
typedef __attribute__((ext_vector_type(8)))  float    v8f;

#define NNODE 50000
#define NEDGE 800000
#define NET   850000          // edges + self-loops
#define NH    6
#define FDIM  64
#define CDIM  384             // NH * FDIM
#define MT    (NNODE / 16)    // 3125
#define NT    (CDIM  / 16)    // 24
#define NTILES (MT * NT)      // 75000

// ---- ordered-uint encoding for float atomic max (identity = 0x00000000) ----
__device__ __forceinline__ unsigned enc_f32(float f) {
  unsigned b = __float_as_uint(f);
  return (b & 0x80000000u) ? ~b : (b | 0x80000000u);
}
__device__ __forceinline__ float dec_f32(unsigned u) {
  return (u & 0x80000000u) ? __uint_as_float(u & 0x7fffffffu)
                           : __uint_as_float(~u);
}

// -------------------------------- zero init ---------------------------------
__global__ void zero_u32(unsigned* __restrict__ p, int n) {
  int t = blockIdx.x * 256 + threadIdx.x;
  if (t < n) p[t] = 0u;
}

// ------------------- Y[N,384] = f16(X[N,64]) @ f16(W[64,384]) ---------------
// One wave per 16x16 tile; K=64 -> two v_wmma_f32_16x16x32_f16.
__global__ __launch_bounds__(256) void gemm_f16_wmma(
    const float* __restrict__ X, const float* __restrict__ W,
    float* __restrict__ Y) {
  const int lane = threadIdx.x & 31;
  const int tile = blockIdx.x * 8 + (threadIdx.x >> 5);
  if (tile >= NTILES) return;                 // wave-uniform: EXEC stays all-1s
  const int m0 = (tile / NT) * 16;
  const int n0 = (tile % NT) * 16;

  // A-matrix (16x32 f16): lane l -> row M=l%16; kb = 8*(l/16)
  const int   mrow = m0 + (lane & 15);
  const int   kb   = (lane >> 4) << 3;
  const float* xr  = X + mrow * FDIM;
  v16h a0, a1, b0, b1;
#pragma unroll
  for (int j = 0; j < 8; ++j) {
    a0[j]     = (_Float16)xr[kb + j];         // K = kb..kb+7
    a0[8 + j] = (_Float16)xr[kb + 16 + j];    // K = kb+16..kb+23
    a1[j]     = (_Float16)xr[kb + 32 + j];    // second WMMA: +32
    a1[8 + j] = (_Float16)xr[kb + 48 + j];
  }
  // B-matrix (32x16 f16): lane l -> col N=l%16; K = kb..kb+7, kb+16..kb+23
  const float* w0 = W + n0 + (lane & 15);
  const float* w1 = w0 + 32 * CDIM;                // second WMMA: +32
#pragma unroll
  for (int j = 0; j < 8; ++j) {
    b0[j]     = (_Float16)w0[(kb + j) * CDIM];
    b0[8 + j] = (_Float16)w0[(kb + 16 + j) * CDIM];
    b1[j]     = (_Float16)w1[(kb + j) * CDIM];
    b1[8 + j] = (_Float16)w1[(kb + 16 + j) * CDIM];
  }

  v8f c = {};
  c = __builtin_amdgcn_wmma_f32_16x16x32_f16(false, a0, false, b0,
                                             (short)0, c, false, false);
  c = __builtin_amdgcn_wmma_f32_16x16x32_f16(false, a1, false, b1,
                                             (short)0, c, false, false);

  // C/D layout: VGPR r, lane l -> M = r + 8*(l/16), N = l%16
  const int ncol  = n0 + (lane & 15);
  const int mbase = m0 + ((lane >> 4) << 3);
#pragma unroll
  for (int r = 0; r < 8; ++r)
    Y[(mbase + r) * CDIM + ncol] = c[r];
}

// ---------------- pass 2: edge attention logits + segment max ---------------
// One wave per edge; lane handles features (l, l+32) of each head.
__global__ __launch_bounds__(256) void edge_scores(
    const int* __restrict__ eidx, const float* __restrict__ gl,
    const float* __restrict__ gr, const float* __restrict__ att,
    float* __restrict__ sc, unsigned* __restrict__ mx) {
  const int lane = threadIdx.x & 31;
  const int e = blockIdx.x * 8 + (threadIdx.x >> 5);
  if (e >= NET) return;
  int src, dst;
  if (e < NEDGE) { src = eidx[e]; dst = eidx[NEDGE + e]; }
  else           { src = dst = e - NEDGE; }

  const float* pl = gl + src * CDIM;
  const float* pr = gr + dst * CDIM;
  float acc[NH];
#pragma unroll
  for (int h = 0; h < NH; ++h) {
    const int b = h * FDIM;
    float u = pl[b + lane]      + pr[b + lane];
    float v = pl[b + 32 + lane] + pr[b + 32 + lane];
    u = u > 0.f ? u : 0.2f * u;               // leaky_relu(0.2)
    v = v > 0.f ? v : 0.2f * v;
    acc[h] = u * att[b + lane] + v * att[b + 32 + lane];
  }
#pragma unroll
  for (int h = 0; h < NH; ++h)
#pragma unroll
    for (int off = 16; off; off >>= 1)
      acc[h] += __shfl_xor(acc[h], off, 32);  // wave32 butterfly

  if (lane == 0) {
#pragma unroll
    for (int h = 0; h < NH; ++h) {
      sc[e * NH + h] = acc[h];
      atomicMax(&mx[dst * NH + h], enc_f32(acc[h]));
    }
  }
}

// ------------- pass 3: p = exp(e - max), segment-sum denominator ------------
__global__ void edge_softmax_num(const int* __restrict__ eidx,
                                 const unsigned* __restrict__ mx,
                                 float* __restrict__ sc,
                                 float* __restrict__ dn) {
  const int t = blockIdx.x * 256 + threadIdx.x;
  if (t >= NET * NH) return;
  const int e = t / NH, h = t - e * NH;
  const int dst = (e < NEDGE) ? eidx[NEDGE + e] : (e - NEDGE);
  const float p = expf(sc[t] - dec_f32(mx[dst * NH + h]));
  sc[t] = p;
  atomicAdd(&dn[dst * NH + h], p);
}

// ------- pass 4: out[dst,f] += sum_h alpha[h] * gl[src, h*64+f] (fold) ------
__global__ __launch_bounds__(256) void aggregate(
    const int* __restrict__ eidx, const float* __restrict__ gl,
    const float* __restrict__ sc, const float* __restrict__ dn,
    float* __restrict__ acc) {
  const int lane = threadIdx.x & 31;
  const int e = blockIdx.x * 8 + (threadIdx.x >> 5);
  if (e >= NET) return;
  int src, dst;
  if (e < NEDGE) { src = eidx[e]; dst = eidx[NEDGE + e]; }
  else           { src = dst = e - NEDGE; }

  float alpha[NH];
#pragma unroll
  for (int h = 0; h < NH; ++h)
    alpha[h] = sc[e * NH + h] / (dn[dst * NH + h] + 1e-16f);

  const float* pl = gl + src * CDIM;
  float s1 = 0.f, s2 = 0.f;
#pragma unroll
  for (int h = 0; h < NH; ++h) {
    s1 += alpha[h] * pl[h * FDIM + lane];
    s2 += alpha[h] * pl[h * FDIM + 32 + lane];
  }
  atomicAdd(&acc[dst * FDIM + lane],      s1);
  atomicAdd(&acc[dst * FDIM + 32 + lane], s2);
}

// ------------------- pass 5: head-mean + bias + SELU ------------------------
__global__ void finalize(const float* __restrict__ acc,
                         const float* __restrict__ bias,
                         float* __restrict__ out) {
  const int t = blockIdx.x * 256 + threadIdx.x;
  if (t >= NNODE * FDIM) return;
  float v = acc[t] * (1.f / (float)NH) + bias[t & (FDIM - 1)];
  const float lam = 1.0507009873554805f, alp = 1.6732632423543772f;
  out[t] = v > 0.f ? lam * v : lam * alp * expm1f(v);
}

extern "C" void kernel_launch(void* const* d_in, const int* in_sizes, int n_in,
                              void* d_out, int out_size, void* d_ws,
                              size_t ws_size, hipStream_t stream) {
  const float* x    = (const float*)d_in[0];
  const int*   eidx = (const int*)d_in[1];     // [2, E] int32
  const float* W_l  = (const float*)d_in[2];   // [64, 384]
  const float* W_r  = (const float*)d_in[3];   // [64, 384]
  const float* att  = (const float*)d_in[4];   // [6, 64]
  const float* bias = (const float*)d_in[5];   // [64]
  float* out = (float*)d_out;

  // workspace layout (f32 words): gl | gr | scores | max(u32) | denom | acc
  float*    gl = (float*)d_ws;
  float*    gr = gl + (size_t)NNODE * CDIM;
  float*    sc = gr + (size_t)NNODE * CDIM;
  unsigned* mx = (unsigned*)(sc + (size_t)NET * NH);
  float*    dn = (float*)(mx + (size_t)NNODE * NH);
  float*    ac = dn + (size_t)NNODE * NH;

  // zero mx + dn + acc (contiguous region)
  const int zn = NNODE * NH * 2 + NNODE * FDIM;
  zero_u32<<<(zn + 255) / 256, 256, 0, stream>>>(mx, zn);

  // dense projections via WMMA
  gemm_f16_wmma<<<NTILES / 8, 256, 0, stream>>>(x, W_l, gl);
  gemm_f16_wmma<<<NTILES / 8, 256, 0, stream>>>(x, W_r, gr);

  // edge phases
  edge_scores<<<(NET + 7) / 8, 256, 0, stream>>>(eidx, gl, gr, att, sc, mx);
  edge_softmax_num<<<(NET * NH + 255) / 256, 256, 0, stream>>>(eidx, mx, sc, dn);
  aggregate<<<(NET + 7) / 8, 256, 0, stream>>>(eidx, gl, sc, dn, ac);

  finalize<<<(NNODE * FDIM + 255) / 256, 256, 0, stream>>>(ac, bias, out);
}
